// Transformer_P_49890340111118
// MI455X (gfx1250) — matrix operand order repaired and checked by reading, not compiled, not run on hardware
//
#include <hip/hip_runtime.h>
#include <math.h>

typedef __attribute__((ext_vector_type(16))) _Float16 v16h;
typedef __attribute__((ext_vector_type(8)))  _Float16 v8h;
typedef __attribute__((ext_vector_type(8)))  float    v8f;

#define D_    512
#define HH    8
#define DHD   64
#define NSEQ  4097
#define NPDD  4352
#define PADR  255
#define MLPH  2048
#define LSUB  17

// ---------------------------------------------------------------------------
// Flexible batched GEMM: C[z] = alpha * A[z] @ B[z] (+bias) (opt ReLU)
//   - f16 WMMA 16x16x32, f32 accumulate; block = 256 thr = 8 wave32
//   - block tile 64M x 128N; each wave computes a 64M x 16N strip: 4 WMMA per
//     K-step sharing one B fragment (amortizes the strided LDS column reads)
//   - REQUIRES: N % 64 == 0, K % 32 == 0, lda/ldb % 8 == 0, aOff % 8 == 0,
//     aLo/aHi % 8 == 0 (all calls below satisfy this) -> every guard is
//     group-uniform and branchless; global traffic is b128.
//   - AMASK + aOff + [aLo,aHi) expresses the shifted-row view used to fold the
//     3-tap expert convs into one K=1536 GEMM.
// ---------------------------------------------------------------------------
struct GemmP {
  const float* A; long lda; long sA; long aOff; long aLo; long aHi;
  const float* B; long ldb; long sB; long bOff;
  float* C; long ldc; long sC;
  const float* bias;
  int M, N, K;
  float alpha;
  int act;
};

template <int AMASK, int TRANSB>
__global__ __launch_bounds__(256) void k_gemm_wmma(GemmP p) {
  __shared__ _Float16 As[64][40];    // [m][k], +8 pad (rows stay 16B aligned)
  __shared__ _Float16 Bs[32][136];   // [k][n], +8 pad
  const int tid = threadIdx.x;
  const int z = blockIdx.z;
  const long m0 = (long)blockIdx.y * 64;
  const long n0 = (long)blockIdx.x * 128;
  const float* A = p.A + (long)z * p.sA;
  const float* B = p.B + (long)z * p.sB + p.bOff;
  float* C = p.C + (long)z * p.sC;

  const int w   = tid >> 5;    // wave id 0..7 = N-strip
  const int ln  = tid & 31;
  const int hl  = ln >> 4;     // lane half
  const int l16 = ln & 15;

  // A staging: 8 contiguous k per thread, one row each (64x32 tile)
  const int ar = tid >> 2, ac = (tid & 3) << 3;
  const long arow = m0 + ar;
  // B staging: 16 contiguous n per thread, one k-row each (32x128 tile)
  const int br = tid >> 3, bc = (tid & 7) << 4;
  const long bcol = n0 + bc;

  v8f acc[4] = {v8f{}, v8f{}, v8f{}, v8f{}};

  for (int k0 = 0; k0 < p.K; k0 += 32) {
    // ---- stage A, branchless group-uniform guard, b128 loads ----
    {
      long gbase = arow * p.lda + (long)(k0 + ac) + p.aOff;
      bool ok = AMASK ? (gbase >= p.aLo && gbase + 8 <= p.aHi)
                      : (arow < p.M);
      const float4* src = (const float4*)(A + (ok ? gbase : 0));
      float4 u0 = src[0], u1 = src[1];
      v8h hv;
      hv[0] = (_Float16)(ok ? u0.x : 0.f); hv[1] = (_Float16)(ok ? u0.y : 0.f);
      hv[2] = (_Float16)(ok ? u0.z : 0.f); hv[3] = (_Float16)(ok ? u0.w : 0.f);
      hv[4] = (_Float16)(ok ? u1.x : 0.f); hv[5] = (_Float16)(ok ? u1.y : 0.f);
      hv[6] = (_Float16)(ok ? u1.z : 0.f); hv[7] = (_Float16)(ok ? u1.w : 0.f);
      *(v8h*)&As[ar][ac] = hv;
    }
    // ---- stage B ----
    {
      bool okb = bcol < p.N;                 // 16-wide group fully in/out
      if constexpr (!TRANSB) {
        long gbase = (long)(k0 + br) * p.ldb + bcol;
        const float4* src = (const float4*)(B + (okb ? gbase : 0));
        float4 u0 = src[0], u1 = src[1], u2 = src[2], u3 = src[3];
        v8h h0, h1;
        h0[0] = (_Float16)(okb ? u0.x : 0.f); h0[1] = (_Float16)(okb ? u0.y : 0.f);
        h0[2] = (_Float16)(okb ? u0.z : 0.f); h0[3] = (_Float16)(okb ? u0.w : 0.f);
        h0[4] = (_Float16)(okb ? u1.x : 0.f); h0[5] = (_Float16)(okb ? u1.y : 0.f);
        h0[6] = (_Float16)(okb ? u1.z : 0.f); h0[7] = (_Float16)(okb ? u1.w : 0.f);
        h1[0] = (_Float16)(okb ? u2.x : 0.f); h1[1] = (_Float16)(okb ? u2.y : 0.f);
        h1[2] = (_Float16)(okb ? u2.z : 0.f); h1[3] = (_Float16)(okb ? u2.w : 0.f);
        h1[4] = (_Float16)(okb ? u3.x : 0.f); h1[5] = (_Float16)(okb ? u3.y : 0.f);
        h1[6] = (_Float16)(okb ? u3.z : 0.f); h1[7] = (_Float16)(okb ? u3.w : 0.f);
        *(v8h*)&Bs[br][bc]     = h0;
        *(v8h*)&Bs[br][bc + 8] = h1;
      } else {
        long cb = okb ? bcol : 0;            // clamped base column (gather)
        v8h h0, h1;
        #pragma unroll
        for (int i = 0; i < 8; ++i) {
          float v0 = B[(cb + i) * p.ldb + (long)(k0 + br)];
          float v1 = B[(cb + 8 + i) * p.ldb + (long)(k0 + br)];
          h0[i] = (_Float16)(okb ? v0 : 0.f);
          h1[i] = (_Float16)(okb ? v1 : 0.f);
        }
        *(v8h*)&Bs[br][bc]     = h0;
        *(v8h*)&Bs[br][bc + 8] = h1;
      }
    }
    if (k0 + 32 < p.K)   // pull next B tile toward L2/L0 (global_prefetch_b8)
      __builtin_prefetch((const void*)&B[TRANSB ? bcol * p.ldb + (k0 + 32)
                                                : (long)(k0 + 32) * p.ldb + bcol], 0, 1);
    __syncthreads();

    // fragments per CDNA5 ISA layouts (16-bit A 16x32, B 32x16)
    v16h af[4], bf;
    #pragma unroll
    for (int e = 0; e < 16; ++e) {
      int j = e >> 1;
      int kk = ((j < 4) ? (j << 1) : 16 + ((j - 4) << 1)) + (hl ? 8 : 0) + (e & 1);
      #pragma unroll
      for (int t = 0; t < 4; ++t) af[t][e] = As[(t << 4) + l16][kk];
      bf[e] = Bs[((e & 8) << 1) + (hl << 3) + (e & 7)][(w << 4) + l16];
    }
    #pragma unroll
    for (int t = 0; t < 4; ++t)
      acc[t] = __builtin_amdgcn_wmma_f32_16x16x32_f16(false, af[t], false, bf,
                                                      (short)0, acc[t], false, false);
    __syncthreads();
  }

  const long col = n0 + (w << 4) + l16;
  if (col < p.N) {
    const float bs = p.bias ? p.bias[col] : 0.f;
    #pragma unroll
    for (int t = 0; t < 4; ++t) {
      #pragma unroll
      for (int r = 0; r < 8; ++r) {
        long row = m0 + (t << 4) + (hl ? 8 : 0) + r;
        if (row < p.M) {
          float o = p.alpha * acc[t][r] + bs;
          if (p.act && o < 0.f) o = 0.f;
          C[row * p.ldc + col] = o;
        }
      }
    }
  }
}

// ---------------------------------------------------------------------------
// Glue kernels
// ---------------------------------------------------------------------------
__global__ __launch_bounds__(256) void k_build_h(const float* f, const float* cls, float* h) {
  long idx = (long)blockIdx.x * 256 + threadIdx.x;
  if (idx >= (long)NSEQ * D_) return;
  int c = idx & 511; long n = idx >> 9;
  h[idx] = (n == 0) ? cls[c] : f[(n - 1) * D_ + c];
}

__global__ __launch_bounds__(256) void k_ln_pad(const float* h, float* x,
                                                const float* g, const float* b) {
  int r = blockIdx.x, t = threadIdx.x;
  if (r < PADR) { x[(long)r * D_ + t] = 0.f; x[(long)r * D_ + t + 256] = 0.f; return; }
  const float* row = h + (long)(r - PADR) * D_;
  float a0 = row[t], a1 = row[t + 256];
  __shared__ float s1[256], s2[256];
  s1[t] = a0 + a1; s2[t] = a0 * a0 + a1 * a1;
  __syncthreads();
  for (int o = 128; o > 0; o >>= 1) { if (t < o) { s1[t] += s1[t + o]; s2[t] += s2[t + o]; } __syncthreads(); }
  float mu = s1[0] * (1.f / D_);
  float var = s2[0] * (1.f / D_) - mu * mu;
  float inv = rsqrtf(var + 1e-5f);
  x[(long)r * D_ + t]       = (a0 - mu) * inv * g[t] + b[t];
  x[(long)r * D_ + t + 256] = (a1 - mu) * inv * g[t + 256] + b[t + 256];
}

__global__ __launch_bounds__(256) void k_softmax(float* x, int ncols) {
  long r = blockIdx.x; int t = threadIdx.x;
  float* row = x + r * (long)ncols;
  __shared__ float red[256];
  float m = -3.4e38f;
  for (int c = t; c < ncols; c += 256) m = fmaxf(m, row[c]);
  red[t] = m; __syncthreads();
  for (int o = 128; o > 0; o >>= 1) { if (t < o) red[t] = fmaxf(red[t], red[t + o]); __syncthreads(); }
  m = red[0]; __syncthreads();
  float s = 0.f;
  for (int c = t; c < ncols; c += 256) { float e = expf(row[c] - m); row[c] = e; s += e; }
  red[t] = s; __syncthreads();
  for (int o = 128; o > 0; o >>= 1) { if (t < o) red[t] += red[t + o]; __syncthreads(); }
  float inv = 1.f / red[0];
  for (int c = t; c < ncols; c += 256) row[c] *= inv;
}

// ql = mean_l(q)*DH^-0.5, kl = mean_l(k) ; landmark pooling
__global__ __launch_bounds__(256) void k_pool(const float* qkv, float* ql, float* kl) {
  long idx = (long)blockIdx.x * 256 + threadIdx.x;
  if (idx >= (long)HH * 256 * DHD) return;
  int d = idx & 63; int m = (idx >> 6) & 255; int h = (int)(idx >> 14);
  float sq = 0.f, sk = 0.f;
  for (int j = 0; j < LSUB; ++j) {
    long n = (long)m * LSUB + j;
    sq += qkv[n * 1536 + h * 64 + d];
    sk += qkv[n * 1536 + 512 + h * 64 + d];
  }
  ql[idx] = sq * (1.f / LSUB) * 0.125f;
  kl[idx] = sk * (1.f / LSUB);
}

__global__ __launch_bounds__(256) void k_pinv_norm(const float* x, float* mr, float* mc) {
  int h = blockIdx.x, t = threadIdx.x;
  const float* X = x + (long)h * 65536;
  float rs = 0.f, cs = 0.f;
  for (int j = 0; j < 256; ++j) { rs += fabsf(X[t * 256 + j]); cs += fabsf(X[j * 256 + t]); }
  __shared__ float r1[256], r2[256];
  r1[t] = rs; r2[t] = cs; __syncthreads();
  for (int o = 128; o > 0; o >>= 1) { if (t < o) { r1[t] = fmaxf(r1[t], r1[t + o]); r2[t] = fmaxf(r2[t], r2[t + o]); } __syncthreads(); }
  if (t == 0) { mr[h] = r1[0]; mc[h] = r2[0]; }
}

__global__ void k_pinv_combine(const float* mr, const float* mc, float* sc) {
  if (threadIdx.x == 0 && blockIdx.x == 0) {
    float a = 0.f, b = 0.f;
    for (int h = 0; h < HH; ++h) { a = fmaxf(a, mr[h]); b = fmaxf(b, mc[h]); }
    sc[0] = 1.f / (a * b);
  }
}

__global__ __launch_bounds__(256) void k_transpose_scale(const float* x, const float* sc, float* z) {
  long idx = (long)blockIdx.x * 256 + threadIdx.x;
  if (idx >= (long)HH * 65536) return;
  long h = idx >> 16; int i = (int)((idx >> 8) & 255); int j = (int)(idx & 255);
  z[idx] = x[(h << 16) + ((long)j << 8) + i] * sc[0];
}

__global__ __launch_bounds__(256) void k_aI_minus_x(float* y, const float* x, float alpha) {
  long idx = (long)blockIdx.x * 256 + threadIdx.x;
  if (idx >= (long)HH * 65536) return;
  int i = (int)((idx >> 8) & 255); int j = (int)(idx & 255);
  y[idx] = ((i == j) ? alpha : 0.f) - x[idx];
}

// depthwise 33-tap conv over sequence on v, added into per-head attn output
__global__ __launch_bounds__(256) void k_resconv(const float* qkv, const float* w, float* attn) {
  long idx = (long)blockIdx.x * 256 + threadIdx.x;
  if (idx >= (long)NPDD * D_) return;
  int hd = idx & 511; int h = hd >> 6; long n = idx >> 9;
  float s = 0.f;
  for (int t = 0; t < 33; ++t) {
    long m = n + t - 16;
    if (m >= 0 && m < NPDD) s += w[h * 33 + t] * qkv[m * 1536 + 1024 + hd];
  }
  attn[idx] += s;
}

__global__ __launch_bounds__(256) void k_add_off(float* h, const float* proj) {
  long idx = (long)blockIdx.x * 256 + threadIdx.x;
  if (idx >= (long)NSEQ * D_) return;
  h[idx] += proj[idx + (long)PADR * D_];
}

// EPEG: depthwise 15-tap conv along image-Y (64x64 grid) on channels; cls kept
__global__ __launch_bounds__(256) void k_epeg(const float* hin, const float* w, const float* b, float* hout) {
  long idx = (long)blockIdx.x * 256 + threadIdx.x;
  if (idx >= (long)NSEQ * D_) return;
  int c = idx & 511; long n = idx >> 9;
  if (n == 0) { hout[idx] = hin[idx]; return; }
  long f = n - 1; int y = (int)(f >> 6); int xx = (int)(f & 63);
  float s = b[c];
  for (int t = 0; t < 15; ++t) {
    int yy = y + t - 7;
    if (yy >= 0 && yy < 64) s += w[c * 15 + t] * hin[(long)(1 + yy * 64 + xx) * D_ + c];
  }
  hout[idx] = s;
}

__global__ __launch_bounds__(256) void k_sigmoid(float* g, const float* x, long total) {
  long idx = (long)blockIdx.x * 256 + threadIdx.x;
  if (idx < total) g[idx] = 1.f / (1.f + expf(-x[idx]));
}

__global__ __launch_bounds__(256) void k_bn(float* t, const float* eb, const float* eg,
                                            const float* ebt, int relu, long total) {
  long idx = (long)blockIdx.x * 256 + threadIdx.x;
  if (idx >= total) return;
  int c = idx & 511;
  float v = (t[idx] + eb[c]) * (eg[c] * rsqrtf(1.f + 1e-5f)) + ebt[c];
  if (relu && v < 0.f) v = 0.f;
  t[idx] = v;
}

__global__ __launch_bounds__(256) void k_gate_add(float* out, const float* g, const float* t, long total) {
  long idx = (long)blockIdx.x * 256 + threadIdx.x;
  if (idx < total) out[idx] += g[idx] * t[idx];
}

// pack ew[e][c][dc][k][1] (c,dc in [0,512), k in [0,3)) -> wp[(k*512+dc)*512 + c]
__global__ __launch_bounds__(256) void k_repack(const float* ew, float* wp) {
  long idx = (long)blockIdx.x * 256 + threadIdx.x;
  if (idx >= (long)1536 * 512) return;
  int c = idx & 511; long kk = idx >> 9; int k = (int)(kk / 512); int dc = (int)(kk % 512);
  wp[idx] = ew[(((long)c * 512 + dc) * 3 + k) * 3 + 1];
}

__global__ __launch_bounds__(256) void k_copy(float* dst, const float* src, long total) {
  long idx = (long)blockIdx.x * 256 + threadIdx.x;
  if (idx < total) dst[idx] = src[idx];
}

// ---------------------------------------------------------------------------
static inline long cdiv(long a, long b) { return (a + b - 1) / b; }

static void gemm(hipStream_t s,
                 const float* A, long lda, long sA, long aOff, long aLo, long aHi, int aMask,
                 const float* B, long ldb, long sB, long bOff, int transB,
                 float* C, long ldc, long sC, const float* bias,
                 int M, int N, int K, float alpha, int act, int batch) {
  GemmP p{A, lda, sA, aOff, aLo, aHi, B, ldb, sB, bOff,
          C, ldc, sC, bias, M, N, K, alpha, act};
  dim3 g((unsigned)cdiv(N, 128), (unsigned)cdiv(M, 64), (unsigned)batch);
  if (aMask)       hipLaunchKernelGGL((k_gemm_wmma<1, 0>), g, dim3(256), 0, s, p);
  else if (transB) hipLaunchKernelGGL((k_gemm_wmma<0, 1>), g, dim3(256), 0, s, p);
  else             hipLaunchKernelGGL((k_gemm_wmma<0, 0>), g, dim3(256), 0, s, p);
}

extern "C" void kernel_launch(void* const* d_in, const int* in_sizes, int n_in,
                              void* d_out, int out_size, void* d_ws, size_t ws_size,
                              hipStream_t stream) {
  const float* features = (const float*)d_in[0];
  const float* cls      = (const float*)d_in[1];
  const float* ln1_g    = (const float*)d_in[2];
  const float* ln1_b    = (const float*)d_in[3];
  const float* qkv1     = (const float*)d_in[4];
  const float* out_w1   = (const float*)d_in[5];
  const float* out_b1   = (const float*)d_in[6];
  const float* res1     = (const float*)d_in[7];
  const float* ln2_g    = (const float*)d_in[8];
  const float* ln2_b    = (const float*)d_in[9];
  const float* qkv2     = (const float*)d_in[10];
  const float* out_w2   = (const float*)d_in[11];
  const float* out_b2   = (const float*)d_in[12];
  const float* res2     = (const float*)d_in[13];
  const float* pe_w     = (const float*)d_in[14];
  const float* pe_b     = (const float*)d_in[15];
  const float* mlp_w1   = (const float*)d_in[16];
  const float* mlp_b1   = (const float*)d_in[17];
  const float* mlp_w2   = (const float*)d_in[18];
  const float* mlp_b2   = (const float*)d_in[19];
  const float* ew1      = (const float*)d_in[20];
  const float* eb1      = (const float*)d_in[21];
  const float* eg1      = (const float*)d_in[22];
  const float* ebt1     = (const float*)d_in[23];
  const float* ew2      = (const float*)d_in[24];
  const float* eb2      = (const float*)d_in[25];
  const float* eg2      = (const float*)d_in[26];
  const float* ebt2     = (const float*)d_in[27];

  // workspace arena
  char* base = (char*)d_ws;
  size_t off = 0;
  auto alloc = [&](size_t elems) -> float* {
    float* p = (float*)(base + off);
    off += ((elems * sizeof(float) + 255) / 256) * 256;
    return p;
  };
  const long HD = (long)NSEQ * D_;            // 4097*512
  float* hA   = alloc(HD);
  float* hB   = alloc(HD);
  float* xln  = alloc((size_t)NPDD * D_);
  float* qkv  = alloc((size_t)NPDD * 1536);
  float* ql   = alloc((size_t)HH * 256 * DHD);
  float* kl   = alloc((size_t)HH * 256 * DHD);
  float* big1 = alloc((size_t)HH * NPDD * 256);   // a1  | reused as mlp hidden
  float* big2 = alloc((size_t)HH * 256 * NPDD);   // a3  | reused as gate,t1,t2
  float* X    = alloc((size_t)HH * 65536);
  float* Za   = alloc((size_t)HH * 65536);
  float* Zb   = alloc((size_t)HH * 65536);
  float* XZ   = alloc((size_t)HH * 65536);
  float* T1   = alloc((size_t)HH * 65536);
  float* T2   = alloc((size_t)HH * 65536);
  float* a3v  = alloc((size_t)HH * 256 * DHD);
  float* pt   = alloc((size_t)HH * 256 * DHD);
  float* attn = alloc((size_t)NPDD * D_);
  float* proj = alloc((size_t)NPDD * D_);
  float* wp   = alloc((size_t)1536 * 512);
  float* mr   = alloc(8); float* mc = alloc(8); float* sc = alloc(4);

  auto launch1d = [&](long total) { return dim3((unsigned)cdiv(total, 256)); };

  // Nystrom attention block: h += attn(ln(h))
  auto nystrom = [&](float* h, const float* qkvW, const float* outW, const float* outB,
                     const float* resW, const float* lng, const float* lnb) {
    hipLaunchKernelGGL(k_ln_pad, dim3(NPDD), dim3(256), 0, stream, h, xln, lng, lnb);
    gemm(stream, xln, D_, 0, 0, 0, 0, 0, qkvW, 1536, 0, 0, 0,
         qkv, 1536, 0, nullptr, NPDD, 1536, D_, 1.f, 0, 1);
    hipLaunchKernelGGL(k_pool, launch1d((long)HH * 256 * DHD), dim3(256), 0, stream, qkv, ql, kl);
    // a1 = softmax(q*s @ klT)
    gemm(stream, qkv, 1536, DHD, 0, 0, 0, 0, kl, DHD, 256 * DHD, 0, 1,
         big1, 256, (long)NPDD * 256, nullptr, NPDD, 256, DHD, 0.125f, 0, HH);
    hipLaunchKernelGGL(k_softmax, dim3(HH * NPDD), dim3(256), 0, stream, big1, 256);
    // a2 = softmax(ql @ klT) -> X
    gemm(stream, ql, DHD, 256 * DHD, 0, 0, 0, 0, kl, DHD, 256 * DHD, 0, 1,
         X, 256, 65536, nullptr, 256, 256, DHD, 1.f, 0, HH);
    hipLaunchKernelGGL(k_softmax, dim3(HH * 256), dim3(256), 0, stream, X, 256);
    // a3 = softmax(ql @ kT)
    gemm(stream, ql, DHD, 256 * DHD, 0, 0, 0, 0, qkv, 1536, DHD, 512, 1,
         big2, NPDD, (long)256 * NPDD, nullptr, 256, NPDD, DHD, 1.f, 0, HH);
    hipLaunchKernelGGL(k_softmax, dim3(HH * 256), dim3(256), 0, stream, big2, NPDD);
    // pinv(X) via Newton-Schulz
    hipLaunchKernelGGL(k_pinv_norm, dim3(HH), dim3(256), 0, stream, X, mr, mc);
    hipLaunchKernelGGL(k_pinv_combine, dim3(1), dim3(32), 0, stream, mr, mc, sc);
    hipLaunchKernelGGL(k_transpose_scale, launch1d((long)HH * 65536), dim3(256), 0, stream, X, sc, Za);
    float* zc = Za; float* zn = Zb;
    for (int it = 0; it < 6; ++it) {
      gemm(stream, X, 256, 65536, 0, 0, 0, 0, zc, 256, 65536, 0, 0,
           XZ, 256, 65536, nullptr, 256, 256, 256, 1.f, 0, HH);
      hipLaunchKernelGGL(k_aI_minus_x, launch1d((long)HH * 65536), dim3(256), 0, stream, T1, XZ, 7.f);
      gemm(stream, XZ, 256, 65536, 0, 0, 0, 0, T1, 256, 65536, 0, 0,
           T2, 256, 65536, nullptr, 256, 256, 256, 1.f, 0, HH);
      hipLaunchKernelGGL(k_aI_minus_x, launch1d((long)HH * 65536), dim3(256), 0, stream, T2, T2, 15.f);
      gemm(stream, XZ, 256, 65536, 0, 0, 0, 0, T2, 256, 65536, 0, 0,
           T1, 256, 65536, nullptr, 256, 256, 256, 1.f, 0, HH);
      hipLaunchKernelGGL(k_aI_minus_x, launch1d((long)HH * 65536), dim3(256), 0, stream, T1, T1, 13.f);
      gemm(stream, zc, 256, 65536, 0, 0, 0, 0, T1, 256, 65536, 0, 0,
           zn, 256, 65536, nullptr, 256, 256, 256, 0.25f, 0, HH);
      float* t = zc; zc = zn; zn = t;
    }
    // a3 @ v
    gemm(stream, big2, NPDD, (long)256 * NPDD, 0, 0, 0, 0, qkv, 1536, DHD, 1024, 0,
         a3v, DHD, 256 * DHD, nullptr, 256, DHD, NPDD, 1.f, 0, HH);
    // pinv @ (a3 v)
    gemm(stream, zc, 256, 65536, 0, 0, 0, 0, a3v, DHD, 256 * DHD, 0, 0,
         pt, DHD, 256 * DHD, nullptr, 256, DHD, 256, 1.f, 0, HH);
    // a1 @ pt -> per-head columns of attn
    gemm(stream, big1, 256, (long)NPDD * 256, 0, 0, 0, 0, pt, DHD, 256 * DHD, 0, 0,
         attn, D_, DHD, nullptr, NPDD, DHD, 256, 1.f, 0, HH);
    hipLaunchKernelGGL(k_resconv, launch1d((long)NPDD * D_), dim3(256), 0, stream, qkv, resW, attn);
    gemm(stream, attn, D_, 0, 0, 0, 0, 0, outW, D_, 0, 0, 0,
         proj, D_, 0, outB, NPDD, D_, D_, 1.f, 0, 1);
    hipLaunchKernelGGL(k_add_off, launch1d(HD), dim3(256), 0, stream, h, proj);
  };

  // PREE: out = relu(x@w1+b1)@w2+b2; 4 sigmoid-gated conv experts
  auto pree = [&](const float* hin, float* hout) {
    float* hid  = big1;                 // [4097,2048]
    float* gate = big2;
    float* t1   = big2 + HD;
    float* t2   = big2 + 2 * HD;
    gemm(stream, hin, D_, 0, 0, 0, 0, 0, mlp_w1, MLPH, 0, 0, 0,
         hid, MLPH, 0, mlp_b1, NSEQ, MLPH, D_, 1.f, 1, 1);
    gemm(stream, hid, MLPH, 0, 0, 0, 0, 0, mlp_w2, D_, 0, 0, 0,
         hout, D_, 0, mlp_b2, NSEQ, D_, MLPH, 1.f, 0, 1);
    hipLaunchKernelGGL(k_sigmoid, launch1d(HD), dim3(256), 0, stream, gate, hin, HD);
    for (int e = 0; e < 4; ++e) {
      const long wstride = (long)512 * 512 * 9;
      hipLaunchKernelGGL(k_repack, launch1d((long)1536 * 512), dim3(256), 0, stream, ew1 + e * wstride, wp);
      gemm(stream, hout, D_, 0, -512, 0, HD, 1, wp, D_, 0, 0, 0,
           t1, D_, 0, nullptr, NSEQ, D_, 1536, 1.f, 0, 1);
      hipLaunchKernelGGL(k_bn, launch1d(HD), dim3(256), 0, stream, t1, eb1 + e * 512, eg1 + e * 512, ebt1 + e * 512, 1, HD);
      hipLaunchKernelGGL(k_repack, launch1d((long)1536 * 512), dim3(256), 0, stream, ew2 + e * wstride, wp);
      gemm(stream, t1, D_, 0, -512, 0, HD, 1, wp, D_, 0, 0, 0,
           t2, D_, 0, nullptr, NSEQ, D_, 1536, 1.f, 0, 1);
      hipLaunchKernelGGL(k_bn, launch1d(HD), dim3(256), 0, stream, t2, eb2 + e * 512, eg2 + e * 512, ebt2 + e * 512, 0, HD);
      hipLaunchKernelGGL(k_gate_add, launch1d(HD), dim3(256), 0, stream, hout, gate, t2, HD);
    }
  };

  // ---- forward ----
  hipLaunchKernelGGL(k_build_h, launch1d(HD), dim3(256), 0, stream, features, cls, hA);

  nystrom(hA, qkv1, out_w1, out_b1, res1, ln1_g, ln1_b);
  hipLaunchKernelGGL(k_epeg, launch1d(HD), dim3(256), 0, stream, hA, pe_w, pe_b, hB);
  pree(hB, hA);

  nystrom(hA, qkv2, out_w2, out_b2, res2, ln2_g, ln2_b);
  hipLaunchKernelGGL(k_epeg, launch1d(HD), dim3(256), 0, stream, hA, pe_w, pe_b, hB);
  pree(hB, hA);

  // output = concat(h[:,0], h[:,1:]) == h flattened
  hipLaunchKernelGGL(k_copy, launch1d(HD), dim3(256), 0, stream, (float*)d_out, hA, HD);
}
